// DGMMLoss_47124381172090
// MI455X (gfx1250) — matrix operand order repaired and checked by reading, not compiled, not run on hardware
//
#include <hip/hip_runtime.h>

typedef __attribute__((ext_vector_type(16))) __bf16 v16bf;
typedef __attribute__((ext_vector_type(8)))  __bf16 v8bf;
typedef __attribute__((ext_vector_type(8)))  float  v8f;

#define N0 4096
#define DD 512
#define NT 8192
#define NCLS 100

// ---------------- prep: mixup, bf16 convert, norms ----------------
__global__ void prep_kernel(const float* __restrict__ x,
                            const float* __restrict__ y,
                            const float* __restrict__ lam_p,
                            const int* __restrict__ perm,
                            float* __restrict__ xul,
                            __bf16* __restrict__ xall,
                            float* __restrict__ nall,
                            float* __restrict__ yall) {
  __shared__ float s1[256], s2[256];
  const int i = blockIdx.x;
  const int t = threadIdx.x;
  const float lam = lam_p[0];
  const int p = perm[i];
  float nx = 0.f, nu = 0.f;
  for (int d = t; d < DD; d += 256) {
    float xv = x[(size_t)i * DD + d];
    float pv = x[(size_t)p * DD + d];
    float uv = lam * xv + (1.0f - lam) * pv;
    xul[(size_t)i * DD + d] = uv;
    __bf16 xb = (__bf16)xv;
    __bf16 ub = (__bf16)uv;
    xall[(size_t)i * DD + d] = xb;
    xall[(size_t)(N0 + i) * DD + d] = ub;
    float xf = (float)xb, uf = (float)ub;
    nx += xf * xf;
    nu += uf * uf;
  }
  s1[t] = nx; s2[t] = nu;
  __syncthreads();
  for (int s = 128; s > 0; s >>= 1) {
    if (t < s) { s1[t] += s1[t + s]; s2[t] += s2[t + s]; }
    __syncthreads();
  }
  if (t == 0) { nall[i] = s1[0]; nall[N0 + i] = s2[0]; yall[i] = y[i]; }
}

// ------- fused pairwise-distance (WMMA bf16) + top-k + mode -------
// wave owns 16 query rows; A fragments (full K=512) cached in VGPRs.
template <int KTOP, bool EXCL>
__global__ void __launch_bounds__(128)
dist_topk_mode(const __bf16* __restrict__ abase,   // xall + qbase*DD
               const __bf16* __restrict__ bbase,   // xall + cbase*DD
               const float* __restrict__ qnorm,    // nall + qbase
               const float* __restrict__ cnorm,    // nall + cbase
               const float* __restrict__ labs_g,   // yall + cbase
               float* __restrict__ mode_out,
               int ntiles) {
  __shared__ float dtile[4 * 256];
  __shared__ float labs_s[16];
  const int wave = threadIdx.x >> 5;
  const int lane = threadIdx.x & 31;
  const int qrow0 = (blockIdx.x * 4 + wave) * 16;
  const int half = lane >> 4;
  const int l15 = lane & 15;

  // A fragments: 16-bit A 16x32 layout — lanes 0-15: K 0..7 & 16..23;
  // lanes 16-31: K 8..15 & 24..31 (ISA 7.12.2). Row = qrow0 + (lane&15).
  const __bf16* arow = abase + (size_t)(qrow0 + l15) * DD;
  v16bf afrag[16];
#pragma unroll
  for (int s = 0; s < 16; ++s) {
    const int k0 = 32 * s + half * 8;
    v8bf lo = *(const v8bf*)(arow + k0);
    v8bf hi = *(const v8bf*)(arow + k0 + 16);
#pragma unroll
    for (int e = 0; e < 8; ++e) { afrag[s][e] = lo[e]; afrag[s][8 + e] = hi[e]; }
  }
  float qn[8];
#pragma unroll
  for (int v = 0; v < 8; ++v) qn[v] = qnorm[qrow0 + v + half * 8];

  float kd[KTOP], kl[KTOP];
#pragma unroll
  for (int q = 0; q < KTOP; ++q) { kd[q] = 3.0e38f; kl[q] = 0.f; }

  for (int tile = 0; tile < ntiles; ++tile) {
    const int j0 = tile * 16;
    v8f acc = {0.f, 0.f, 0.f, 0.f, 0.f, 0.f, 0.f, 0.f};
    // B 32x16 layout: lane = column (l15), same K order as A (lanes 16-31: K 8..15 & 24..31).
    const __bf16* bcol = bbase + (size_t)(j0 + l15) * DD + half * 8;
#pragma unroll
    for (int s = 0; s < 16; ++s) {
      v8bf blo = *(const v8bf*)(bcol + 32 * s);
      v8bf bhi = *(const v8bf*)(bcol + 32 * s + 16);
      v16bf bf;
#pragma unroll
      for (int e = 0; e < 8; ++e) { bf[e] = blo[e]; bf[8 + e] = bhi[e]; }
      acc = __builtin_amdgcn_wmma_f32_16x16x32_bf16(false, afrag[s], false, bf,
                                                    (short)0, acc, false, false);
    }
    const float cn = cnorm[j0 + l15];
    __syncthreads();  // previous tile's update phase done
    if (threadIdx.x < 16) labs_s[threadIdx.x] = labs_g[j0 + threadIdx.x];
#pragma unroll
    for (int v = 0; v < 8; ++v) {
      const int m = v + half * 8;  // C/D layout: lanes16-31 -> M = v+8
      dtile[wave * 256 + m * 16 + l15] = qn[v] + cn - 2.0f * acc[v];
    }
    __syncthreads();
    if (lane < 16) {
      const int qi = qrow0 + lane;
#pragma unroll
      for (int n = 0; n < 16; ++n) {
        const int j = j0 + n;
        const float dd = dtile[wave * 256 + lane * 16 + n];
        if (EXCL && j == qi) continue;
        if (dd < kd[KTOP - 1]) {  // sorted ascending; replace max + bubble
          kd[KTOP - 1] = dd;
          kl[KTOP - 1] = labs_s[n];
#pragma unroll
          for (int q = KTOP - 1; q > 0; --q) {
            if (kd[q] < kd[q - 1]) {
              float td = kd[q]; kd[q] = kd[q - 1]; kd[q - 1] = td;
              float tl = kl[q]; kl[q] = kl[q - 1]; kl[q - 1] = tl;
            }
          }
        }
      }
    }
  }

  if (lane < 16) {  // mode: most frequent, ties -> smallest value
    float bestv = 3.0e38f;
    int bestc = -1;
#pragma unroll
    for (int a = 0; a < KTOP; ++a) {
      int cnt = 0;
#pragma unroll
      for (int b = 0; b < KTOP; ++b) cnt += (kl[b] == kl[a]) ? 1 : 0;
      if (cnt > bestc || (cnt == bestc && kl[a] < bestv)) { bestc = cnt; bestv = kl[a]; }
    }
    mode_out[qrow0 + lane] = bestv;
  }
}

// ---------------- per-class means (f32, no atomics) ----------------
__global__ void class_means(const float* __restrict__ x,
                            const float* __restrict__ xul,
                            const float* __restrict__ yall,
                            float* __restrict__ mu,
                            float* __restrict__ counts) {
  __shared__ float slab[512];
  const int c = blockIdx.x;
  const int t = threadIdx.x;  // dim
  const float cf = (float)c;
  float acc = 0.f;
  int cnt = 0;
  for (int base = 0; base < NT; base += 512) {
    __syncthreads();
    slab[t] = yall[base + t];
    __syncthreads();
    for (int ii = 0; ii < 512; ++ii) {
      if (slab[ii] == cf) {
        ++cnt;
        const int r = base + ii;
        acc += (r < N0) ? x[(size_t)r * DD + t] : xul[(size_t)(r - N0) * DD + t];
      }
    }
  }
  mu[(size_t)c * DD + t] = acc / fmaxf((float)cnt, 1.f);
  if (t == 0) counts[c] = (float)cnt;
}

// ---------------- GM loss per row ----------------
__global__ void __launch_bounds__(128)
gm_loss(const float* __restrict__ x, const float* __restrict__ xul,
        const float* __restrict__ yall, const float* __restrict__ mu,
        const float* __restrict__ counts, float* __restrict__ gm_acc) {
  __shared__ float row[DD];
  __shared__ float red[128];
  const int i = blockIdx.x;
  const int t = threadIdx.x;
  const float* src = (i < N0) ? (x + (size_t)i * DD) : (xul + (size_t)(i - N0) * DD);
  for (int d = t; d < DD; d += 128) row[d] = src[d];
  __syncthreads();
  float pi = 0.f;
  if (t < NCLS) {
    float d2 = 0.f;
    const float* m = mu + (size_t)t * DD;
    for (int d = 0; d < DD; ++d) { float df = row[d] - m[d]; d2 += df * df; }
    pi = __expf(-0.5f * d2) * (counts[t] > 0.f ? 1.f : 0.f);
  }
  red[t] = pi;
  __syncthreads();
  for (int s = 64; s > 0; s >>= 1) { if (t < s) red[t] += red[t + s]; __syncthreads(); }
  const float pisum = red[0] + 1e-15f;
  __syncthreads();
  float contrib = 0.f;
  if (t < NCLS) {
    float pv = fminf(fmaxf(pi / pisum, 0.f), 1.f);
    float yh = ((int)yall[i] == t) ? 1.f : 0.f;
    float df = pv - yh;
    contrib = df * df;
  }
  red[t] = contrib;
  __syncthreads();
  for (int s = 64; s > 0; s >>= 1) { if (t < s) red[t] += red[t + s]; __syncthreads(); }
  if (t == 0) atomicAdd(gm_acc, red[0]);
}

__global__ void init_acc(float* gm_acc) { if (threadIdx.x == 0) gm_acc[0] = 0.f; }

__global__ void finalize_kernel(const float* __restrict__ modes2,
                                const float* __restrict__ yall,
                                const float* __restrict__ gm_acc,
                                float* __restrict__ out) {
  __shared__ float red[256];
  const int t = threadIdx.x;
  float s = 0.f;
  for (int i = t; i < NT; i += 256) { float d = modes2[i] - yall[i]; s += d * d; }
  red[t] = s;
  __syncthreads();
  for (int k = 128; k > 0; k >>= 1) { if (t < k) red[t] += red[t + k]; __syncthreads(); }
  if (t == 0) out[0] = gm_acc[0] / (float)NT + 0.01f * (red[0] / (float)NT);
}

extern "C" void kernel_launch(void* const* d_in, const int* in_sizes, int n_in,
                              void* d_out, int out_size, void* d_ws, size_t ws_size,
                              hipStream_t stream) {
  (void)in_sizes; (void)n_in; (void)out_size; (void)ws_size;
  const float* x = (const float*)d_in[0];
  const float* y = (const float*)d_in[1];
  const float* lam = (const float*)d_in[2];
  const int* perm = (const int*)d_in[3];

  char* ws = (char*)d_ws;
  size_t off = 0;
  float* xul = (float*)(ws + off);   off += (size_t)N0 * DD * 4;   // 8 MB
  __bf16* xall = (__bf16*)(ws + off); off += (size_t)NT * DD * 2;  // 8 MB
  float* nall = (float*)(ws + off);  off += (size_t)NT * 4;
  float* yall = (float*)(ws + off);  off += (size_t)NT * 4;
  float* mu = (float*)(ws + off);    off += (size_t)NCLS * DD * 4;
  float* counts = (float*)(ws + off); off += 512;
  float* modes2 = (float*)(ws + off); off += (size_t)NT * 4;
  float* gm_acc = (float*)(ws + off); off += 256;

  init_acc<<<1, 32, 0, stream>>>(gm_acc);
  prep_kernel<<<N0, 256, 0, stream>>>(x, y, lam, perm, xul, xall, nall, yall);
  // branch 1: queries = x_ul rows [N0,2N0), candidates = x rows [0,N0), k=11
  dist_topk_mode<11, false><<<N0 / 64, 128, 0, stream>>>(
      xall + (size_t)N0 * DD, xall, nall + N0, nall, yall, yall + N0, N0 / 16);
  // branch 2: full 8192x8192, k=3, exclude self
  dist_topk_mode<3, true><<<NT / 64, 128, 0, stream>>>(
      xall, xall, nall, nall, yall, modes2, NT / 16);
  class_means<<<NCLS, 512, 0, stream>>>(x, xul, yall, mu, counts);
  gm_loss<<<NT, 128, 0, stream>>>(x, xul, yall, mu, counts, gm_acc);
  finalize_kernel<<<1, 256, 0, stream>>>(modes2, yall, gm_acc, (float*)d_out);
}
